// DeepSeekV2MoE_1563368096322
// MI455X (gfx1250) — matrix operand order repaired and checked by reading, not compiled, not run on hardware
//
#include <hip/hip_runtime.h>
#include <hip/hip_bf16.h>

typedef __attribute__((ext_vector_type(16))) __bf16 v16bf;
typedef __attribute__((ext_vector_type(8)))  __bf16 v8bf;
typedef __attribute__((ext_vector_type(4)))  __bf16 v4bf;
typedef __attribute__((ext_vector_type(8)))  float  v8f;

#define T_TOK 8192
#define HDIM  2048
#define IDIM  1408
#define NEXP  16
#define CAP   2048      // per-expert slot capacity (mean 1024, std ~31 -> safe)
#define BM    128
#define BN    32
#define BK    32
#define LDP   (BK + 8)  // LDS row pitch in bf16 (80B = 20 banks -> conflict-free)

__device__ __forceinline__ v16bf frag_combine(v8bf lo, v8bf hi) {
    v16bf r;
#pragma unroll
    for (int i = 0; i < 8; ++i) { r[i] = lo[i]; r[8 + i] = hi[i]; }
    return r;
}

// async global->LDS 16B copy (CDNA5), tracked by ASYNCcnt
__device__ __forceinline__ void async_copy_b128(unsigned ldsOff, const void* gptr) {
    asm volatile("global_load_async_to_lds_b128 %0, %1, off"
                 :: "v"(ldsOff), "v"(gptr) : "memory");
}
__device__ __forceinline__ void async_wait0() {
    asm volatile("s_wait_asynccnt 0" ::: "memory");
}

// ---------------- init: zero output + expert counters ----------------
__global__ void moe_init(float* __restrict__ out, int* __restrict__ cnt, size_t n) {
    size_t i = (size_t)blockIdx.x * blockDim.x + threadIdx.x;
    size_t stride = (size_t)gridDim.x * blockDim.x;
    for (size_t j = i; j < n; j += stride) out[j] = 0.0f;
    if (i < NEXP) cnt[i] = 0;
}

// ---------------- x fp32 -> bf16 ----------------
__global__ void moe_cvt_x(const float* __restrict__ x, __bf16* __restrict__ xb, int n4) {
    int i = blockIdx.x * blockDim.x + threadIdx.x;
    if (i < n4) {
        float4 f = ((const float4*)x)[i];
        v4bf b;
        b[0] = (__bf16)f.x; b[1] = (__bf16)f.y; b[2] = (__bf16)f.z; b[3] = (__bf16)f.w;
        ((v4bf*)xb)[i] = b;
    }
}

// ---------------- gating: logits, top-2, softmax, build expert lists ----------------
__global__ __launch_bounds__(256) void moe_gate(
    const float* __restrict__ x, const float* __restrict__ gw,
    int* __restrict__ cnt, int* __restrict__ tok, float* __restrict__ wts) {
    const int lane = threadIdx.x & 31;
    const int wave = threadIdx.x >> 5;
    const int t = blockIdx.x * 8 + wave;
    if (t >= T_TOK) return;

    float acc[NEXP];
#pragma unroll
    for (int e = 0; e < NEXP; ++e) acc[e] = 0.0f;

    for (int k = lane; k < HDIM; k += 32) {
        float xv = x[(size_t)t * HDIM + k];
#pragma unroll
        for (int e = 0; e < NEXP; ++e) acc[e] += xv * gw[(size_t)e * HDIM + k];
    }
#pragma unroll
    for (int e = 0; e < NEXP; ++e) {
        float v = acc[e];
        for (int o = 16; o >= 1; o >>= 1) v += __shfl_xor(v, o, 32);
        acc[e] = v;
    }
    if (lane == 0) {
        int b0 = 0; float m0 = acc[0];
        for (int e = 1; e < NEXP; ++e) if (acc[e] > m0) { m0 = acc[e]; b0 = e; }
        int b1 = (b0 == 0) ? 1 : 0; float m1 = acc[b1];
        for (int e = 0; e < NEXP; ++e)
            if (e != b0 && acc[e] > m1) { m1 = acc[e]; b1 = e; }
        float w0 = 1.0f / (1.0f + __expf(m1 - m0));   // softmax over the two selected
        float w1 = 1.0f - w0;
        int s0 = atomicAdd(&cnt[b0], 1);
        if (s0 < CAP) { tok[b0 * CAP + s0] = t; wts[b0 * CAP + s0] = w0; }
        int s1 = atomicAdd(&cnt[b1], 1);
        if (s1 < CAP) { tok[b1 * CAP + s1] = t; wts[b1 * CAP + s1] = w1; }
    }
}

// ---------------- grouped GEMM1: h = silu(x W1^T) * (x Up^T), bf16 out ----------------
// Block 256 thr (8 waves). Tile BM=128 x BN=32, BK=32.
// Wave (mPair, nSub): computes M-subtiles 2*mPair, 2*mPair+1 sharing B frags.
__global__ __launch_bounds__(256) void moe_gemm1(
    const __bf16* __restrict__ xb, const float* __restrict__ W1,
    const float* __restrict__ Up, const int* __restrict__ cnt,
    const int* __restrict__ tok, __bf16* __restrict__ hbuf) {
    const int e = blockIdx.z;
    int count = cnt[e]; if (count > CAP) count = CAP;
    const int row0 = blockIdx.y * BM;
    if (row0 >= count) return;
    const int n0 = blockIdx.x * BN;

    __shared__ __bf16 sA[BM][LDP];
    __shared__ __bf16 sW[BN][LDP];
    __shared__ __bf16 sU[BN][LDP];

    const int tid = threadIdx.x, lane = tid & 31, wave = tid >> 5;
    const int mPair = wave & 3, nSub = wave >> 2;

    // A staging: 128 rows x 32 K bf16; each thread async-copies rows ar, ar+64
    const int ar = tid >> 2, ac = (tid & 3) * 8;
    int g0 = row0 + ar;       if (g0 > count - 1) g0 = count - 1;
    int g1 = row0 + ar + 64;  if (g1 > count - 1) g1 = count - 1;
    const __bf16* aP0 = xb + (size_t)tok[e * CAP + g0] * HDIM + ac;
    const __bf16* aP1 = xb + (size_t)tok[e * CAP + g1] * HDIM + ac;
    const unsigned ldsA0 = (unsigned)(uintptr_t)&sA[ar][ac];
    const unsigned ldsA1 = (unsigned)(uintptr_t)&sA[ar + 64][ac];

    // B staging: 32 rows x 32 K fp32 -> bf16, 4 elems per thread per matrix
    const int br = tid >> 3, bc = (tid & 7) * 4;
    const float* w1p = W1 + ((size_t)e * IDIM + (n0 + br)) * HDIM + bc;
    const float* upp = Up + ((size_t)e * IDIM + (n0 + br)) * HDIM + bc;

    v8f acc1[2] = {}; v8f accu[2] = {};

    for (int k0 = 0; k0 < HDIM; k0 += BK) {
        // async A tile (issued right after previous trailing barrier -> no race)
        async_copy_b128(ldsA0, aP0 + k0);
        async_copy_b128(ldsA1, aP1 + k0);
        // B tiles: fp32 load, convert, stage
        float4 f1 = *(const float4*)(w1p + k0);
        float4 fu = *(const float4*)(upp + k0);
        v4bf c1; c1[0] = (__bf16)f1.x; c1[1] = (__bf16)f1.y; c1[2] = (__bf16)f1.z; c1[3] = (__bf16)f1.w;
        *(v4bf*)&sW[br][bc] = c1;
        v4bf cu; cu[0] = (__bf16)fu.x; cu[1] = (__bf16)fu.y; cu[2] = (__bf16)fu.z; cu[3] = (__bf16)fu.w;
        *(v4bf*)&sU[br][bc] = cu;
        async_wait0();
        __syncthreads();

        const int kb = (lane >> 4) * 8;   // A: K halves kb..kb+7, kb+16..kb+23
        const int ks = (lane >> 4) * 8;   // B: K halves ks..ks+7, ks+16..ks+23
        const __bf16* a0 = &sA[(mPair * 2 + 0) * 16 + (lane & 15)][0];
        const __bf16* a1 = &sA[(mPair * 2 + 1) * 16 + (lane & 15)][0];
        v16bf aF0 = frag_combine(*(const v8bf*)(a0 + kb), *(const v8bf*)(a0 + kb + 16));
        v16bf aF1 = frag_combine(*(const v8bf*)(a1 + kb), *(const v8bf*)(a1 + kb + 16));
        const __bf16* wB = &sW[nSub * 16 + (lane & 15)][0];
        v16bf bW = frag_combine(*(const v8bf*)(wB + ks), *(const v8bf*)(wB + ks + 16));
        const __bf16* uB = &sU[nSub * 16 + (lane & 15)][0];
        v16bf bU = frag_combine(*(const v8bf*)(uB + ks), *(const v8bf*)(uB + ks + 16));

        acc1[0] = __builtin_amdgcn_wmma_f32_16x16x32_bf16(false, aF0, false, bW, (short)0, acc1[0], false, false);
        acc1[1] = __builtin_amdgcn_wmma_f32_16x16x32_bf16(false, aF1, false, bW, (short)0, acc1[1], false, false);
        accu[0] = __builtin_amdgcn_wmma_f32_16x16x32_bf16(false, aF0, false, bU, (short)0, accu[0], false, false);
        accu[1] = __builtin_amdgcn_wmma_f32_16x16x32_bf16(false, aF1, false, bU, (short)0, accu[1], false, false);
        __syncthreads();
    }

    const int mOff = (lane >> 4) * 8;
    const int col = n0 + nSub * 16 + (lane & 15);
#pragma unroll
    for (int s = 0; s < 2; ++s) {
        const int mBase = row0 + (mPair * 2 + s) * 16 + mOff;
#pragma unroll
        for (int j = 0; j < 8; ++j) {
            int gm = mBase + j;
            if (gm < count) {
                float a1 = acc1[s][j];
                float hv = (a1 / (1.0f + __expf(-a1))) * accu[s][j];   // silu(a1)*up
                hbuf[((size_t)e * CAP + gm) * (size_t)IDIM + col] = (__bf16)hv;
            }
        }
    }
}

// ---------------- grouped GEMM2: out += we * (h W2^T), atomic scatter ----------------
__global__ __launch_bounds__(256) void moe_gemm2(
    const __bf16* __restrict__ hbuf, const float* __restrict__ W2,
    const int* __restrict__ cnt, const int* __restrict__ tok,
    const float* __restrict__ wts, float* __restrict__ out) {
    const int e = blockIdx.z;
    int count = cnt[e]; if (count > CAP) count = CAP;
    const int row0 = blockIdx.y * BM;
    if (row0 >= count) return;
    const int n0 = blockIdx.x * BN;

    __shared__ __bf16 sA[BM][LDP];
    __shared__ __bf16 sW[BN][LDP];

    const int tid = threadIdx.x, lane = tid & 31, wave = tid >> 5;
    const int mPair = wave & 3, nSub = wave >> 2;

    const int ar = tid >> 2, ac = (tid & 3) * 8;
    int g0 = row0 + ar;       if (g0 > count - 1) g0 = count - 1;
    int g1 = row0 + ar + 64;  if (g1 > count - 1) g1 = count - 1;
    const __bf16* aP0 = hbuf + ((size_t)e * CAP + g0) * (size_t)IDIM + ac;
    const __bf16* aP1 = hbuf + ((size_t)e * CAP + g1) * (size_t)IDIM + ac;
    const unsigned ldsA0 = (unsigned)(uintptr_t)&sA[ar][ac];
    const unsigned ldsA1 = (unsigned)(uintptr_t)&sA[ar + 64][ac];

    const int br = tid >> 3, bc = (tid & 7) * 4;
    const float* w2p = W2 + ((size_t)e * HDIM + (n0 + br)) * IDIM + bc;

    v8f acc[2] = {};
    for (int k0 = 0; k0 < IDIM; k0 += BK) {
        async_copy_b128(ldsA0, aP0 + k0);
        async_copy_b128(ldsA1, aP1 + k0);
        float4 f2 = *(const float4*)(w2p + k0);
        v4bf c2; c2[0] = (__bf16)f2.x; c2[1] = (__bf16)f2.y; c2[2] = (__bf16)f2.z; c2[3] = (__bf16)f2.w;
        *(v4bf*)&sW[br][bc] = c2;
        async_wait0();
        __syncthreads();

        const int kb = (lane >> 4) * 8;
        const int ks = (lane >> 4) * 8;
        const __bf16* a0 = &sA[(mPair * 2 + 0) * 16 + (lane & 15)][0];
        const __bf16* a1 = &sA[(mPair * 2 + 1) * 16 + (lane & 15)][0];
        v16bf aF0 = frag_combine(*(const v8bf*)(a0 + kb), *(const v8bf*)(a0 + kb + 16));
        v16bf aF1 = frag_combine(*(const v8bf*)(a1 + kb), *(const v8bf*)(a1 + kb + 16));
        const __bf16* wB = &sW[nSub * 16 + (lane & 15)][0];
        v16bf bW = frag_combine(*(const v8bf*)(wB + ks), *(const v8bf*)(wB + ks + 16));

        acc[0] = __builtin_amdgcn_wmma_f32_16x16x32_bf16(false, aF0, false, bW, (short)0, acc[0], false, false);
        acc[1] = __builtin_amdgcn_wmma_f32_16x16x32_bf16(false, aF1, false, bW, (short)0, acc[1], false, false);
        __syncthreads();
    }

    const int mOff = (lane >> 4) * 8;
    const int col = n0 + nSub * 16 + (lane & 15);
#pragma unroll
    for (int s = 0; s < 2; ++s) {
        const int mBase = row0 + (mPair * 2 + s) * 16 + mOff;
#pragma unroll
        for (int j = 0; j < 8; ++j) {
            int gm = mBase + j;
            if (gm < count) {
                int   tv = tok[e * CAP + gm];
                float we = wts[e * CAP + gm];
                atomicAdd(out + (size_t)tv * HDIM + col, acc[s][j] * we);
            }
        }
    }
}

// ---------------- launcher ----------------
extern "C" void kernel_launch(void* const* d_in, const int* in_sizes, int n_in,
                              void* d_out, int out_size, void* d_ws, size_t ws_size,
                              hipStream_t stream) {
    (void)in_sizes; (void)n_in; (void)out_size; (void)ws_size;
    const float* x  = (const float*)d_in[0];
    const float* gw = (const float*)d_in[1];
    const float* W1 = (const float*)d_in[2];
    const float* Up = (const float*)d_in[3];
    const float* W2 = (const float*)d_in[4];
    float* out = (float*)d_out;

    // workspace layout (~126 MB)
    char* ws = (char*)d_ws;
    __bf16* xb   = (__bf16*)ws;                                  // T*H*2   = 33,554,432
    size_t  off  = (size_t)T_TOK * HDIM * 2;
    int*    cnt  = (int*)(ws + off);          off += 256;        // E counters (padded)
    int*    tok  = (int*)(ws + off);          off += (size_t)NEXP * CAP * 4;
    float*  wts  = (float*)(ws + off);        off += (size_t)NEXP * CAP * 4;
    __bf16* hbuf = (__bf16*)(ws + off);                          // E*CAP*I*2 = 92,274,688

    // 1) zero out + counters
    moe_init<<<2048, 256, 0, stream>>>(out, cnt, (size_t)T_TOK * HDIM);
    // 2) x -> bf16
    const int n4 = T_TOK * HDIM / 4;
    moe_cvt_x<<<n4 / 256, 256, 0, stream>>>(x, xb, n4);
    // 3) gating (8 tokens per 256-thread block, one wave32 per token)
    moe_gate<<<T_TOK / 8, 256, 0, stream>>>(x, gw, cnt, tok, wts);
    // 4) GEMM1: grid = (I/BN, CAP/BM, E); empty row-tiles early-exit
    moe_gemm1<<<dim3(IDIM / BN, CAP / BM, NEXP), 256, 0, stream>>>(xb, W1, Up, cnt, tok, hbuf);
    // 5) GEMM2: grid = (H/BN, CAP/BM, E)
    moe_gemm2<<<dim3(HDIM / BN, CAP / BM, NEXP), 256, 0, stream>>>(hbuf, W2, cnt, tok, wts, out);
}
